// HMCModel_30691836297906
// MI455X (gfx1250) — matrix operand order repaired and checked by reading, not compiled, not run on hardware
//
#include <hip/hip_runtime.h>
#include <hip/hip_bf16.h>

#define H  30
#define HS 32          // padded feature stride (K=32 for WMMA)
#define SLOPE 0.2f

typedef __attribute__((ext_vector_type(16))) __bf16 v16bf;
typedef __attribute__((ext_vector_type(8)))  float  v8f;

// ---------------------------------------------------------------------------
// Pad raw input X[n x D] -> Xp[npad x 32] (zeros beyond n rows / D cols)
// ---------------------------------------------------------------------------
__global__ void k_pad_input(const float* __restrict__ X, int n, int D,
                            float* __restrict__ Xp, int npad)
{
    int idx = blockIdx.x * blockDim.x + threadIdx.x;
    int r = idx >> 5, c = idx & 31;
    if (r >= npad) return;
    Xp[(size_t)r * HS + c] = (r < n && c < D) ? X[(size_t)r * D + c] : 0.0f;
}

// Prep weight: W[Kdim x 30] row-major f32 -> Wb[32 x 32] bf16, layout [c*32+k].
// Cols 0..29 = W^T; cols 30/31 = folded attention vectors W@a[:H], W@a[H:]
// (so the GEMM produces per-node scalars p,q for free), or 0 if avec==null.
__global__ void k_prep_w(const float* __restrict__ W, int Kdim,
                         const float* __restrict__ avec, __bf16* __restrict__ Wb)
{
    int tid = threadIdx.x;              // 1024 threads
    int c = tid >> 5, k = tid & 31;
    float v = 0.0f;
    if (k < Kdim) {
        if (c < H) {
            v = W[k * H + c];
        } else if (avec) {
            const float* ah = avec + (c == 31 ? H : 0);
            float s = 0.0f;
            for (int h = 0; h < H; ++h) s += W[k * H + h] * ah[h];
            v = s;
        }
    }
    Wb[c * 32 + k] = (__bf16)v;
}

// Prep bias: b[30] (or none) -> out[32] zero-padded
__global__ void k_prep_bias(const float* __restrict__ b, int has, float* __restrict__ out)
{
    int c = threadIdx.x;                // 32 threads
    out[c] = (has && c < H) ? b[c] : 0.0f;
}

// ---------------------------------------------------------------------------
// WMMA GEMM: OUT[npad x 32] = Xp[npad x 32] @ Wb (+bias32). One wave per
// 16-row slab; two v_wmma_f32_16x16x32_bf16. Branch-free hot path.
// A layout (16-bit 16x32): lane L, row M=L%16, a[i] -> K=g*16+i
// B layout (16-bit 32x16): lane L col N=L%16, b[i] -> K=g*16+i
// C layout (f32 16x16):    VGPR v, lane L -> row v+g*8, col L%16
// ---------------------------------------------------------------------------
__global__ void __launch_bounds__(32)
k_gemm_wmma(const float* __restrict__ Xp, const __bf16* __restrict__ Wb,
            const float* __restrict__ bias32, float* __restrict__ OUT)
{
    const int L    = threadIdx.x;       // 0..31, wave32
    const int row0 = blockIdx.x * 16;
    const int M    = L & 15;
    const int g    = L >> 4;

    const float4* rp = (const float4*)(Xp + (size_t)(row0 + M) * HS);
    float4 fa = rp[4 * g];              // K = g*16   .. g*16+3
    float4 fb = rp[4 * g + 1];          // K = g*16+4 .. g*16+7
    float4 fc = rp[4 * g + 2];          // K = g*16+8 ..
    float4 fd = rp[4 * g + 3];

    v16bf a;
    a[0]  = (__bf16)fa.x; a[1]  = (__bf16)fa.y; a[2]  = (__bf16)fa.z; a[3]  = (__bf16)fa.w;
    a[4]  = (__bf16)fb.x; a[5]  = (__bf16)fb.y; a[6]  = (__bf16)fb.z; a[7]  = (__bf16)fb.w;
    a[8]  = (__bf16)fc.x; a[9]  = (__bf16)fc.y; a[10] = (__bf16)fc.z; a[11] = (__bf16)fc.w;
    a[12] = (__bf16)fd.x; a[13] = (__bf16)fd.y; a[14] = (__bf16)fd.z; a[15] = (__bf16)fd.w;

    const int c0 = M, c1 = M + 16;
    v16bf b0 = *(const v16bf*)(Wb + c0 * 32 + g * 16);
    v16bf b1 = *(const v16bf*)(Wb + c1 * 32 + g * 16);

    v8f acc0 = {}, acc1 = {};
    acc0 = __builtin_amdgcn_wmma_f32_16x16x32_bf16(false, a, false, b0, (short)0, acc0, false, false);
    acc1 = __builtin_amdgcn_wmma_f32_16x16x32_bf16(false, a, false, b1, (short)0, acc1, false, false);

    const float bi0 = bias32[c0];
    const float bi1 = bias32[c1];       // 0 for pad cols
#pragma unroll
    for (int v = 0; v < 8; ++v) {
        const size_t rr = (size_t)(row0 + v + g * 8) * HS;
        OUT[rr + c0] = acc0[v] + bi0;
        OUT[rr + c1] = acc1[v] + bi1;
    }
}

__global__ void k_fill(float* __restrict__ ptr, float v, size_t cnt)
{
    size_t i = (size_t)blockIdx.x * blockDim.x + threadIdx.x;
    if (i < cnt) ptr[i] = v;
}

__device__ __forceinline__ float lrelu(float x) { return x > 0.f ? x : SLOPE * x; }

// e = lrelu(P[ip*32] + Q[iq*32]); D[is] += e.  P/Q are feature-matrix bases
// pre-offset by +30 (p) / +31 (q): attention scalars live in pad cols of M.
__global__ void k_edge_denom(const float* __restrict__ P, const int* __restrict__ ip,
                             const float* __restrict__ Q, const int* __restrict__ iq,
                             const int* __restrict__ is, float* __restrict__ D, int nnz)
{
    int t = blockIdx.x * blockDim.x + threadIdx.x;
    if (t >= nnz) return;
    float e = lrelu(P[(size_t)ip[t] * HS] + Q[(size_t)iq[t] * HS]);
    atomicAdd(&D[is[t]], e);
}

// OUT[is]*32+h += (e/D[is]) * vals * F[ifx]*32+h   (32 lanes per edge, h<30)
__global__ void k_edge_scatter(const float* __restrict__ P, const int* __restrict__ ip,
                               const float* __restrict__ Q, const int* __restrict__ iq,
                               const int* __restrict__ is, const float* __restrict__ D,
                               const float* __restrict__ vals,
                               const float* __restrict__ F, const int* __restrict__ ifx,
                               float* __restrict__ OUT, int nnz)
{
    long long gid = (long long)blockIdx.x * blockDim.x + threadIdx.x;
    // t is wave-uniform (wave32, 256-thread blocks): force scalar path
    int t = __builtin_amdgcn_readfirstlane((int)(gid >> 5));
    int h = (int)(gid & 31);
    if (t >= nnz || h >= H) return;
    int s = is[t];
    float att = lrelu(P[(size_t)ip[t] * HS] + Q[(size_t)iq[t] * HS]) / D[s];
    if (vals) att *= vals[t];
    atomicAdd(&OUT[(size_t)s * HS + h], att * F[(size_t)ifx[t] * HS + h]);
}

__global__ void k_acc_relu_add(float* __restrict__ ACC, const float* __restrict__ T, size_t cnt)
{
    size_t i = (size_t)blockIdx.x * blockDim.x + threadIdx.x;
    if (i < cnt) ACC[i] += fmaxf(T[i], 0.f);
}

__global__ void k_relu_copy(float* __restrict__ dst, const float* __restrict__ src, size_t cnt)
{
    size_t i = (size_t)blockIdx.x * blockDim.x + threadIdx.x;
    if (i < cnt) dst[i] = fmaxf(src[i], 0.f);
}

// head: per-node dot with w (30); nanmean bookkeeping (sum of non-NaN, count)
__global__ void k_head(const float* __restrict__ Xf, const float* __restrict__ w,
                       float* __restrict__ sum, float* __restrict__ cnt, int n)
{
    int r = blockIdx.x * blockDim.x + threadIdx.x;
    if (r >= n) return;
    const float* row = Xf + (size_t)r * HS;
    float s = 0.f;
#pragma unroll
    for (int h = 0; h < H; ++h) s += row[h] * w[h];
    if (!isnan(s)) { atomicAdd(sum, s); atomicAdd(cnt, 1.0f); }
}

__global__ void k_final(const float* __restrict__ hacc,
                        const float* __restrict__ b0, const float* __restrict__ b1,
                        const float* __restrict__ b2, float* __restrict__ out)
{
    if (blockIdx.x == 0 && threadIdx.x == 0) {
        out[0] = hacc[0] / hacc[1] + b0[0]
               + hacc[2] / hacc[3] + b1[0]
               + hacc[4] / hacc[5] + b2[0];
    }
}

// ---------------------------------------------------------------------------
// Host-side orchestration
// ---------------------------------------------------------------------------
struct Rank { float *CUR, *ACC, *T, *M, *d; int n, npad; };

static inline void fillz(float* p, size_t cnt, hipStream_t s)
{
    k_fill<<<(unsigned)((cnt + 255) / 256), 256, 0, s>>>(p, 0.f, cnt);
}

// GEMM with on-the-fly weight prep; avec!=null folds p,q into out cols 30/31
static inline void gemm(const float* Xp, int Kdim, const float* Wf, const float* avec,
                        const float* bias32, __bf16* Wb, float* OUT, int npad, hipStream_t s)
{
    k_prep_w<<<1, 1024, 0, s>>>(Wf, Kdim, avec, Wb);
    k_gemm_wmma<<<npad / 16, 32, 0, s>>>(Xp, Wb, bias32, OUT);
}

// HBS block: m|p|q = CUR@[W | W@a]; e = lrelu(p[rows]+q[cols]); row-norm;
// ACC += relu(seg(att*m[cols]))
static void hbs(Rank& R, const int* rows, const int* cols, int nnz,
                const float* Wm, const float* a, __bf16* Wb, const float* zb32,
                hipStream_t s)
{
    const size_t S = (size_t)R.npad * HS;
    gemm(R.CUR, H, Wm, a, zb32, Wb, R.M, R.npad, s);
    const float* p = R.M + 30; const float* q = R.M + 31;
    fillz(R.d, R.n, s);
    k_edge_denom<<<(nnz + 255) / 256, 256, 0, s>>>(p, rows, q, cols, rows, R.d, nnz);
    fillz(R.T, S, s);
    k_edge_scatter<<<(unsigned)(((long long)nnz * 32 + 255) / 256), 256, 0, s>>>(
        p, rows, q, cols, rows, R.d, nullptr, R.M, cols, R.T, nnz);
    k_acc_relu_add<<<(unsigned)((S + 255) / 256), 256, 0, s>>>(R.ACC, R.T, S);
}

// HBNS block between source rank Rs and target rank Rt
static void hbns(Rank& Rs, Rank& Rt, const int* rows_t, const int* cols_s,
                 const float* vals, int nnz, const float* ws_, const float* wt_,
                 const float* a, __bf16* Wb, const float* zb32, hipStream_t s)
{
    const size_t Ss = (size_t)Rs.npad * HS, St = (size_t)Rt.npad * HS;
    gemm(Rs.CUR, H, ws_, a, zb32, Wb, Rs.M, Rs.npad, s);   // s_msg | p_s | q_s
    gemm(Rt.CUR, H, wt_, a, zb32, Wb, Rt.M, Rt.npad, s);   // t_msg | p_t | q_t
    const float* ps = Rs.M + 30; const float* qs = Rs.M + 31;
    const float* pt = Rt.M + 30; const float* qt = Rt.M + 31;
    // e_st = lrelu(p_s[cols] + q_t[rows]); normalized over rows_t
    fillz(Rt.d, Rt.n, s);
    k_edge_denom<<<(nnz + 255) / 256, 256, 0, s>>>(ps, cols_s, qt, rows_t, rows_t, Rt.d, nnz);
    // e_ts = lrelu(p_t[rows] + q_s[cols]); normalized over cols_s
    fillz(Rs.d, Rs.n, s);
    k_edge_denom<<<(nnz + 255) / 256, 256, 0, s>>>(pt, rows_t, qs, cols_s, cols_s, Rs.d, nnz);
    // msg_t = seg(att_st*vals*s_msg[cols], rows_t) -> ACC_t += relu(.)
    fillz(Rt.T, St, s);
    k_edge_scatter<<<(unsigned)(((long long)nnz * 32 + 255) / 256), 256, 0, s>>>(
        ps, cols_s, qt, rows_t, rows_t, Rt.d, vals, Rs.M, cols_s, Rt.T, nnz);
    k_acc_relu_add<<<(unsigned)((St + 255) / 256), 256, 0, s>>>(Rt.ACC, Rt.T, St);
    // msg_s = seg(att_ts*vals*t_msg[rows], cols_s) -> ACC_s += relu(.)
    fillz(Rs.T, Ss, s);
    k_edge_scatter<<<(unsigned)(((long long)nnz * 32 + 255) / 256), 256, 0, s>>>(
        pt, rows_t, qs, cols_s, cols_s, Rs.d, vals, Rt.M, rows_t, Rs.T, nnz);
    k_acc_relu_add<<<(unsigned)((Ss + 255) / 256), 256, 0, s>>>(Rs.ACC, Rs.T, Ss);
}

extern "C" void kernel_launch(void* const* d_in, const int* in_sizes, int n_in,
                              void* d_out, int out_size, void* d_ws, size_t ws_size,
                              hipStream_t stream)
{
    (void)n_in; (void)out_size; (void)ws_size;
    const float* x0 = (const float*)d_in[0];
    const float* x1 = (const float*)d_in[1];
    const float* x2 = (const float*)d_in[2];
    const int* a0r = (const int*)d_in[3];  const int* a0c = (const int*)d_in[4];
    const int* a1r = (const int*)d_in[6];  const int* a1c = (const int*)d_in[7];
    const int* c2r = (const int*)d_in[9];  const int* c2c = (const int*)d_in[10];
    const int* i1r = (const int*)d_in[12]; const int* i1c = (const int*)d_in[13];
    const float* i1v = (const float*)d_in[14];
    const int* i2r = (const int*)d_in[15]; const int* i2c = (const int*)d_in[16];
    const float* i2v = (const float*)d_in[17];
    // params dict flattened in insertion order (matches _make_params definition)
    const float* P[32];
    for (int i = 0; i < 32; ++i) P[i] = (const float*)d_in[18 + i];
    // 0:lin0_in_w 1:lin0_in_b 2:lin1_in_w 3:lin1_in_b 4:lin2_in_w 5:lin2_in_b
    // 6:hbs0_l1_w 7:hbs0_l1_a 8:hbns01_l1_ws 9:hbns01_l1_wt 10:hbns01_l1_a
    // 11:hbns12_l1_ws 12:hbns12_l1_wt 13:hbns12_l1_a
    // 14:hbs0_l2_w 15:hbs0_l2_a 16:hbs1_l2_w 17:hbs1_l2_a 18:hbs2_l2_w 19:hbs2_l2_a
    // 20:hbns01_l2_ws 21:hbns01_l2_wt 22:hbns01_l2_a 23:hbns12_l2_ws 24:hbns12_l2_wt 25:hbns12_l2_a
    // 26:lin0_w 27:lin0_b 28:lin1_w 29:lin1_b 30:lin2_w 31:lin2_b

    const int N0 = in_sizes[0] / 14, N1 = in_sizes[1] / 6, N2 = in_sizes[2] / 10;
    const int NNZ_A0 = in_sizes[3], NNZ_A1 = in_sizes[6], NNZ_C2 = in_sizes[9];
    const int NNZ_I1 = in_sizes[12], NNZ_I2 = in_sizes[15];
    const int NP0 = ((N0 + 15) / 16) * 16, NP1 = ((N1 + 15) / 16) * 16, NP2 = ((N2 + 15) / 16) * 16;

    // workspace carve-out (all block sizes are multiples of 16 floats => >=64B aligned)
    float* w = (float*)d_ws;
    auto alloc = [&](size_t cnt) { float* p = w; w += cnt; return p; };
    Rank R0, R1, R2;
    R0.n = N0; R0.npad = NP0; R1.n = N1; R1.npad = NP1; R2.n = N2; R2.npad = NP2;
    const size_t S0 = (size_t)NP0 * HS, S1 = (size_t)NP1 * HS, S2 = (size_t)NP2 * HS;
    R0.CUR = alloc(S0); R0.ACC = alloc(S0); R0.T = alloc(S0); R0.M = alloc(S0); R0.d = alloc(NP0);
    R1.CUR = alloc(S1); R1.ACC = alloc(S1); R1.T = alloc(S1); R1.M = alloc(S1); R1.d = alloc(NP1);
    R2.CUR = alloc(S2); R2.ACC = alloc(S2); R2.T = alloc(S2); R2.M = alloc(S2); R2.d = alloc(NP2);
    float* X0p = alloc(S0); float* X1p = alloc(S1); float* X2p = alloc(S2);
    __bf16* Wb   = (__bf16*)alloc(512);   // 32x32 bf16 staging tile
    float*  b32  = alloc(32);             // staged bias
    float*  zb32 = alloc(32);             // zero bias
    float*  hacc = alloc(16);             // sum0,cnt0,sum1,cnt1,sum2,cnt2

    // ---- pad raw inputs, prep zero bias ----
    k_pad_input<<<(unsigned)(((size_t)NP0 * 32 + 255) / 256), 256, 0, stream>>>(x0, N0, 14, X0p, NP0);
    k_pad_input<<<(unsigned)(((size_t)NP1 * 32 + 255) / 256), 256, 0, stream>>>(x1, N1, 6,  X1p, NP1);
    k_pad_input<<<(unsigned)(((size_t)NP2 * 32 + 255) / 256), 256, 0, stream>>>(x2, N2, 10, X2p, NP2);
    k_prep_bias<<<1, 32, 0, stream>>>(nullptr, 0, zb32);

    // ---- input linears (WMMA) ----
    k_prep_bias<<<1, 32, 0, stream>>>(P[1], 1, b32);
    gemm(X0p, 14, P[0], nullptr, b32, Wb, R0.CUR, NP0, stream);
    k_prep_bias<<<1, 32, 0, stream>>>(P[3], 1, b32);
    gemm(X1p, 6,  P[2], nullptr, b32, Wb, R1.CUR, NP1, stream);
    k_prep_bias<<<1, 32, 0, stream>>>(P[5], 1, b32);
    gemm(X2p, 10, P[4], nullptr, b32, Wb, R2.CUR, NP2, stream);

    // ---- HMC layer, level 1 ----
    fillz(R0.ACC, S0, stream); fillz(R1.ACC, S1, stream); fillz(R2.ACC, S2, stream);
    hbs(R0, a0r, a0c, NNZ_A0, P[6], P[7], Wb, zb32, stream);
    hbns(R1, R0, i1r, i1c, i1v, NNZ_I1, P[8],  P[9],  P[10], Wb, zb32, stream);
    hbns(R2, R1, i2r, i2c, i2v, NNZ_I2, P[11], P[12], P[13], Wb, zb32, stream);
    k_relu_copy<<<(unsigned)((S0 + 255) / 256), 256, 0, stream>>>(R0.CUR, R0.ACC, S0);
    k_relu_copy<<<(unsigned)((S1 + 255) / 256), 256, 0, stream>>>(R1.CUR, R1.ACC, S1);
    k_relu_copy<<<(unsigned)((S2 + 255) / 256), 256, 0, stream>>>(R2.CUR, R2.ACC, S2);

    // ---- HMC layer, level 2 ----
    fillz(R0.ACC, S0, stream); fillz(R1.ACC, S1, stream); fillz(R2.ACC, S2, stream);
    hbs(R0, a0r, a0c, NNZ_A0, P[14], P[15], Wb, zb32, stream);
    hbs(R1, a1r, a1c, NNZ_A1, P[16], P[17], Wb, zb32, stream);
    hbs(R2, c2r, c2c, NNZ_C2, P[18], P[19], Wb, zb32, stream);
    hbns(R1, R0, i1r, i1c, i1v, NNZ_I1, P[20], P[21], P[22], Wb, zb32, stream);
    hbns(R2, R1, i2r, i2c, i2v, NNZ_I2, P[23], P[24], P[25], Wb, zb32, stream);
    k_relu_copy<<<(unsigned)((S0 + 255) / 256), 256, 0, stream>>>(R0.CUR, R0.ACC, S0);
    k_relu_copy<<<(unsigned)((S1 + 255) / 256), 256, 0, stream>>>(R1.CUR, R1.ACC, S1);
    k_relu_copy<<<(unsigned)((S2 + 255) / 256), 256, 0, stream>>>(R2.CUR, R2.ACC, S2);

    // ---- heads + nanmean pooling ----
    fillz(hacc, 16, stream);
    k_head<<<(N0 + 255) / 256, 256, 0, stream>>>(R0.CUR, P[26], &hacc[0], &hacc[1], N0);
    k_head<<<(N1 + 255) / 256, 256, 0, stream>>>(R1.CUR, P[28], &hacc[2], &hacc[3], N1);
    k_head<<<(N2 + 255) / 256, 256, 0, stream>>>(R2.CUR, P[30], &hacc[4], &hacc[5], N2);
    k_final<<<1, 32, 0, stream>>>(hacc, P[27], P[29], P[31], (float*)d_out);
}
